// SpectralFirstMambaAutoencoderV2_3753801417444
// MI455X (gfx1250) — compile-verified, matrix operand order checked by reading, not run on hardware
//
#include <hip/hip_runtime.h>
#include <math.h>

// ---------------------------------------------------------------------------
// CDNA5 WMMA types
// ---------------------------------------------------------------------------
typedef __attribute__((ext_vector_type(16))) _Float16 v16h;
typedef __attribute__((ext_vector_type(8)))  float    v8f;

#define ACT_NONE 0
#define ACT_GELU 1

#define CEILDIV(a,b) (((a)+(b)-1)/(b))

__device__ __forceinline__ float gelu_f(float x) {
    // exact gelu: 0.5*x*(1+erf(x/sqrt(2)))
    return 0.5f * x * (1.0f + erff(x * 0.70710678118654752f));
}
__device__ __forceinline__ float silu_f(float x) {
    return x / (1.0f + __expf(-x));
}
__device__ __forceinline__ float softplus_f(float x) {
    return (x > 20.0f) ? x : log1pf(__expf(x));
}
__device__ __forceinline__ float apply_act(float v, int act) {
    return (act == ACT_GELU) ? gelu_f(v) : v;
}

// ---------------------------------------------------------------------------
// WMMA GEMM:  C[M,N] (+)= act( A[M,lda(:K)] @ W[N,K]^T + bias[N] )
// One wave computes one 16x16 tile; 4 waves (n-tiles) per 128-thread block.
// K is a template parameter (64 / 128) so the K-loop fully unrolls into
// b128 load clauses + v_cvt_pk_f16_f32 + v_wmma_f32_16x16x32_f16.
// Out-of-range columns are handled by CLAMPING the weight-row pointer
// (no exec divergence; WMMA requires EXEC all ones anyway); stores guarded.
// Requires: M % 16 == 0, K % 32 == 0, 16B alignment of A/W rows.
// ---------------------------------------------------------------------------
template<int K>
__global__ __launch_bounds__(128) void gemm_wmma_k(
    const float* __restrict__ A, const float* __restrict__ W,
    const float* __restrict__ bias, float* __restrict__ C,
    int N, int lda, int ldc, int act, int accum)
{
    const int lane  = threadIdx.x;            // 0..31 (wave32)
    const int mloc  = lane & 15;
    const int half  = lane >> 4;              // K-group selector
    const int m0    = blockIdx.x * 16;
    const int tn    = blockIdx.y * 4 + threadIdx.y;
    const int ncol  = tn * 16 + mloc;
    const int ncl   = (ncol < N) ? ncol : (N - 1);   // clamp: no divergence

    const float* abase = A + (size_t)(m0 + mloc) * lda + half * 8;
    const float* wbase = W + (size_t)ncl * K + half * 8;

    v8f acc = {};
#pragma unroll
    for (int k0 = 0; k0 < K; k0 += 32) {
        // A fragment: 16-bit A 16x32 layout (ISA 7.12.2): per lane two runs
        // of 8 consecutive K values at offsets half*8 and 16+half*8.
        float4 a0 = *(const float4*)(abase + k0);
        float4 a1 = *(const float4*)(abase + k0 + 4);
        float4 a2 = *(const float4*)(abase + k0 + 16);
        float4 a3 = *(const float4*)(abase + k0 + 20);
        float4 b0 = *(const float4*)(wbase + k0);
        float4 b1 = *(const float4*)(wbase + k0 + 4);
        float4 b2 = *(const float4*)(wbase + k0 + 16);
        float4 b3 = *(const float4*)(wbase + k0 + 20);

        float aarr[16] = { a0.x,a0.y,a0.z,a0.w, a1.x,a1.y,a1.z,a1.w,
                           a2.x,a2.y,a2.z,a2.w, a3.x,a3.y,a3.z,a3.w };
        float barr[16] = { b0.x,b0.y,b0.z,b0.w, b1.x,b1.y,b1.z,b1.w,
                           b2.x,b2.y,b2.z,b2.w, b3.x,b3.y,b3.z,b3.w };
        v16h af, bf;
#pragma unroll
        for (int j = 0; j < 16; ++j) {
            af[j] = (_Float16)aarr[j];
            bf[j] = (_Float16)barr[j];
        }
        // D = A x B + C (f32 accum); EXEC uniform all-ones here.
        acc = __builtin_amdgcn_wmma_f32_16x16x32_f16(
            /*neg_a=*/false, af, /*neg_b=*/false, bf,
            /*c_mod=*/(short)0, acc, /*reuse_a=*/false, /*reuse_b=*/false);
    }

    if (ncol < N) {
        const float bv = bias ? bias[ncol] : 0.0f;
#pragma unroll
        for (int r = 0; r < 8; ++r) {
            const int m = m0 + r + half * 8;   // C/D layout: VGPR r -> row r + 8*half
            float v = apply_act(acc[r] + bv, act);
            const size_t idx = (size_t)m * ldc + ncol;
            if (accum) C[idx] += v; else C[idx] = v;
        }
    }
}

// ---------------------------------------------------------------------------
// Generic direct conv2d, NCHW, square kernel.
// ---------------------------------------------------------------------------
__global__ __launch_bounds__(256) void conv2d_k(
    const float* __restrict__ in, const float* __restrict__ w,
    const float* __restrict__ bias, float* __restrict__ out,
    int Cin, int Hin, int Win, int Cout, int Hout, int Wout,
    int Kk, int stride, int pad, int act)
{
    int idx = blockIdx.x * blockDim.x + threadIdx.x;
    int total = Cout * Hout * Wout;
    if (idx >= total) return;
    int ox = idx % Wout; int t = idx / Wout;
    int oy = t % Hout;   int oc = t / Hout;
    float s = bias ? bias[oc] : 0.0f;
    for (int ic = 0; ic < Cin; ++ic) {
        const float* ip = in + (size_t)ic * Hin * Win;
        const float* wp = w + ((size_t)oc * Cin + ic) * Kk * Kk;
        for (int ky = 0; ky < Kk; ++ky) {
            int iy = oy * stride - pad + ky;
            if (iy < 0 || iy >= Hin) continue;
            for (int kx = 0; kx < Kk; ++kx) {
                int ix = ox * stride - pad + kx;
                if (ix < 0 || ix >= Win) continue;
                s += ip[iy * Win + ix] * wp[ky * Kk + kx];
            }
        }
    }
    out[idx] = apply_act(s, act);
}

// ---------------------------------------------------------------------------
// Transposed conv (kernel 4, stride 2, effective pad 1 -> Hout = 2*Hin).
// w layout: [Cin, Cout, 4, 4]  (matches reference convt2d).
// ---------------------------------------------------------------------------
__global__ __launch_bounds__(256) void convt_k(
    const float* __restrict__ in, const float* __restrict__ w,
    const float* __restrict__ bias, float* __restrict__ out,
    int Cin, int Hin, int Win, int Cout, int act)
{
    int Hout = 2 * Hin, Wout = 2 * Win;
    int idx = blockIdx.x * blockDim.x + threadIdx.x;
    int total = Cout * Hout * Wout;
    if (idx >= total) return;
    int ox = idx % Wout; int t = idx / Wout;
    int oy = t % Hout;   int oc = t / Hout;
    float s = bias[oc];
    for (int ic = 0; ic < Cin; ++ic) {
        const float* ip = in + (size_t)ic * Hin * Win;
        const float* wp = w + ((size_t)ic * Cout + oc) * 16;
        for (int j = 0; j < 4; ++j) {
            int iy2 = oy + 1 - j;
            if (iy2 < 0 || (iy2 & 1)) continue;
            int iy = iy2 >> 1;
            if (iy >= Hin) continue;
            for (int k = 0; k < 4; ++k) {
                int ix2 = ox + 1 - k;
                if (ix2 < 0 || (ix2 & 1)) continue;
                int ix = ix2 >> 1;
                if (ix >= Win) continue;
                s += ip[iy * Win + ix] * wp[j * 4 + k];
            }
        }
    }
    out[idx] = apply_act(s, act);
}

// ---------------------------------------------------------------------------
// 2x2 average pool of x (64,64,64) -> xlow flat [n*64 + c], n = i*32 + j.
// ---------------------------------------------------------------------------
__global__ __launch_bounds__(256) void avgpool_k(const float* __restrict__ x,
                                                 float* __restrict__ xlow)
{
    int idx = blockIdx.x * blockDim.x + threadIdx.x;
    if (idx >= 1024 * 64) return;
    int c = idx & 63; int n = idx >> 6;
    int i = n >> 5;   int j = n & 31;
    const float* p = x + (size_t)c * 64 * 64;
    float v = p[(2*i)*64 + 2*j] + p[(2*i)*64 + 2*j+1]
            + p[(2*i+1)*64 + 2*j] + p[(2*i+1)*64 + 2*j+1];
    xlow[idx] = 0.25f * v;
}

// htok[m,d] = xlow[m] * te_w[d] + te_b[d] + pos[t,d],  m = n*64+t
__global__ __launch_bounds__(256) void embed_k(
    const float* __restrict__ xlow, const float* __restrict__ te_w,
    const float* __restrict__ te_b, const float* __restrict__ pos,
    float* __restrict__ htok)
{
    size_t e = (size_t)blockIdx.x * blockDim.x + threadIdx.x;
    if (e >= (size_t)65536 * 64) return;
    int d = (int)(e & 63);
    size_t m = e >> 6;
    int t = (int)(m & 63);
    htok[e] = xlow[m] * te_w[d] + te_b[d] + pos[t * 64 + d];
}

// LayerNorm over last dim = 64, one thread per row, float4 loads/stores.
__global__ __launch_bounds__(256) void ln_k(
    const float* __restrict__ in, const float* __restrict__ g,
    const float* __restrict__ b, float* __restrict__ out, int M)
{
    int r = blockIdx.x * blockDim.x + threadIdx.x;
    if (r >= M) return;
    const float4* x4 = (const float4*)(in + (size_t)r * 64);
    float xs[64];
    float sum = 0.0f, sq = 0.0f;
#pragma unroll
    for (int i = 0; i < 16; ++i) {
        float4 t = x4[i];
        xs[4*i+0] = t.x; xs[4*i+1] = t.y; xs[4*i+2] = t.z; xs[4*i+3] = t.w;
        sum += t.x + t.y + t.z + t.w;
        sq  += t.x*t.x + t.y*t.y + t.z*t.z + t.w*t.w;
    }
    float m = sum * (1.0f / 64.0f);
    float v = sq * (1.0f / 64.0f) - m * m;
    float inv = rsqrtf(v + 1e-5f);
    float4* o4 = (float4*)(out + (size_t)r * 64);
#pragma unroll
    for (int i = 0; i < 16; ++i) {
        float4 t;
        t.x = (xs[4*i+0] - m) * inv * g[4*i+0] + b[4*i+0];
        t.y = (xs[4*i+1] - m) * inv * g[4*i+1] + b[4*i+1];
        t.z = (xs[4*i+2] - m) * inv * g[4*i+2] + b[4*i+2];
        t.w = (xs[4*i+3] - m) * inv * g[4*i+3] + b[4*i+3];
        o4[i] = t;
    }
}

// Depthwise causal conv1d (K=4) over t, + bias, + SiLU.
// Input = xz[:, 0:128]; rev=1 uses the reversed-sequence window t..t+3.
__global__ __launch_bounds__(256) void dwconv_silu_k(
    const float* __restrict__ xz, const float* __restrict__ w,
    const float* __restrict__ b, float* __restrict__ xcc, int rev)
{
    size_t idx = (size_t)blockIdx.x * blockDim.x + threadIdx.x;
    if (idx >= (size_t)65536 * 128) return;
    int c = (int)(idx & 127);
    size_t m = idx >> 7;
    int t = (int)(m & 63);
    size_t n = m >> 6;
    float s = b[c];
#pragma unroll
    for (int j = 0; j < 4; ++j) {
        int ts = rev ? (t + 3 - j) : (t - 3 + j);
        if (ts >= 0 && ts < 64)
            s += w[c * 4 + j] * xz[(n * 64 + (size_t)ts) * 256 + c];
    }
    xcc[m * 128 + c] = silu_f(s);
}

// dt[m,c] = softplus( dbl[m,0:4] @ dt_w[c,0:4] + dt_b[c] )
__global__ __launch_bounds__(256) void dtproj_k(
    const float* __restrict__ dbl, const float* __restrict__ dt_w,
    const float* __restrict__ dt_b, float* __restrict__ dtb)
{
    size_t idx = (size_t)blockIdx.x * blockDim.x + threadIdx.x;
    if (idx >= (size_t)65536 * 128) return;
    int c = (int)(idx & 127);
    size_t m = idx >> 7;
    const float* dr = dbl + m * 36;
    float s = dt_b[c];
#pragma unroll
    for (int r = 0; r < 4; ++r) s += dr[r] * dt_w[c * 4 + r];
    dtb[m * 128 + c] = softplus_f(s);
}

// Selective-scan: one thread per (sequence n, channel d); 16-state SSM in regs.
// a, bu recomputed on the fly; y gated by silu(z). rev=1 scans t=63..0.
__global__ __launch_bounds__(256) void scan_k(
    const float* __restrict__ dbl, const float* __restrict__ dtb,
    const float* __restrict__ xcc, const float* __restrict__ xz,
    const float* __restrict__ A_log, const float* __restrict__ Dp,
    float* __restrict__ ybuf, int rev)
{
    int idx = blockIdx.x * blockDim.x + threadIdx.x;
    if (idx >= 1024 * 128) return;
    int d = idx & 127;
    int n = idx >> 7;
    float Arow[16], h[16];
#pragma unroll
    for (int s = 0; s < 16; ++s) { Arow[s] = -__expf(A_log[d * 16 + s]); h[s] = 0.0f; }
    const float Dd = Dp[d];
    for (int step = 0; step < 64; ++step) {
        int t = rev ? (63 - step) : step;
        size_t m = (size_t)n * 64 + t;
        float dt = dtb[m * 128 + d];
        float x  = xcc[m * 128 + d];
        const float* dr = dbl + m * 36;
        float acc = 0.0f;
#pragma unroll
        for (int s = 0; s < 16; ++s) {
            float a = __expf(dt * Arow[s]);
            h[s] = a * h[s] + (dt * dr[4 + s]) * x;
            acc += h[s] * dr[20 + s];
        }
        float y = acc + x * Dd;
        float z = xz[m * 256 + 128 + d];
        ybuf[m * 128 + d] = y * silu_f(z);
    }
}

// Softmax attention pooling per sequence n: block of 64 threads.
__global__ __launch_bounds__(64) void softmax_pool_k(
    const float* __restrict__ h, const float* __restrict__ sw,
    float* __restrict__ pooled)
{
    __shared__ float al[64];
    __shared__ float red[64];
    int n = blockIdx.x, t = threadIdx.x;
    const float* row = h + ((size_t)n * 64 + t) * 64;
    float s = 0.0f;
    for (int d = 0; d < 64; ++d) s += row[d] * sw[d];
    red[t] = s; __syncthreads();
    for (int o = 32; o > 0; o >>= 1) { if (t < o) red[t] = fmaxf(red[t], red[t + o]); __syncthreads(); }
    float mx = red[0]; __syncthreads();
    float e = __expf(s - mx);
    al[t] = e; red[t] = e; __syncthreads();
    for (int o = 32; o > 0; o >>= 1) { if (t < o) red[t] += red[t + o]; __syncthreads(); }
    float denom = red[0]; __syncthreads();
    // thread t now acts as channel d
    float acc = 0.0f;
    for (int tt = 0; tt < 64; ++tt)
        acc += al[tt] * h[((size_t)n * 64 + tt) * 64 + t];
    pooled[(size_t)n * 64 + t] = acc / denom;
}

// feat [n,96] -> NCHW (96,32,32)
__global__ __launch_bounds__(256) void feat2chw_k(const float* __restrict__ feat,
                                                  float* __restrict__ fchw)
{
    int idx = blockIdx.x * blockDim.x + threadIdx.x;
    if (idx >= 96 * 1024) return;
    int n = idx & 1023; int c = idx >> 10;
    fchw[c * 1024 + n] = feat[(size_t)n * 96 + c];
}

// FiLM: spec = spec * (1 + gamma) + beta  (elementwise, 96x16x16)
__global__ __launch_bounds__(256) void film_k(float* __restrict__ spec,
                                              const float* __restrict__ gamma,
                                              const float* __restrict__ beta)
{
    int idx = blockIdx.x * blockDim.x + threadIdx.x;
    if (idx >= 96 * 16 * 16) return;
    spec[idx] = spec[idx] * (1.0f + gamma[idx]) + beta[idx];
}

// zc3 NCHW (64,64,64) -> sp[p*64 + c], p = y*64 + x
__global__ __launch_bounds__(256) void chw2sp_k(const float* __restrict__ zc3,
                                                float* __restrict__ sp)
{
    int idx = blockIdx.x * blockDim.x + threadIdx.x;
    if (idx >= 64 * 64 * 64) return;
    int c = idx & 63; int p = idx >> 6;
    sp[idx] = zc3[(size_t)c * 4096 + p];
}

// Refinement block: per-row residual 1->16->16->1 conv1d chain along c (len 64).
__global__ __launch_bounds__(64) void ref_k(
    float* __restrict__ sp,
    const float* __restrict__ w1, const float* __restrict__ b1,
    const float* __restrict__ w2, const float* __restrict__ b2,
    const float* __restrict__ w3, const float* __restrict__ b3)
{
    __shared__ float u0[66];
    __shared__ float h1[16][66];
    __shared__ float h2[16][66];
    int p = blockIdx.x, c = threadIdx.x;
    float v = sp[(size_t)p * 64 + c];
    u0[c + 1] = v;
    if (c == 0) { u0[0] = 0.0f; u0[65] = 0.0f; }
    __syncthreads();
    for (int r = 0; r < 16; ++r) {
        float s = b1[r] + w1[r*3+0]*u0[c] + w1[r*3+1]*u0[c+1] + w1[r*3+2]*u0[c+2];
        h1[r][c + 1] = gelu_f(s);
    }
    if (c == 0) for (int r = 0; r < 16; ++r) { h1[r][0] = 0.0f; h1[r][65] = 0.0f; }
    __syncthreads();
    for (int r = 0; r < 16; ++r) {
        float s = b2[r];
        for (int q = 0; q < 16; ++q) {
            const float* wq = w2 + ((size_t)r * 16 + q) * 3;
            s += wq[0]*h1[q][c] + wq[1]*h1[q][c+1] + wq[2]*h1[q][c+2];
        }
        h2[r][c + 1] = gelu_f(s);
    }
    if (c == 0) for (int r = 0; r < 16; ++r) { h2[r][0] = 0.0f; h2[r][65] = 0.0f; }
    __syncthreads();
    float s = b3[0];
    for (int q = 0; q < 16; ++q)
        s += w3[q*3+0]*h2[q][c] + w3[q*3+1]*h2[q][c+1] + w3[q*3+2]*h2[q][c+2];
    sp[(size_t)p * 64 + c] = v + s;
}

// out NCHW = sigmoid(sp transposed)
__global__ __launch_bounds__(256) void final_k(const float* __restrict__ sp,
                                               float* __restrict__ out)
{
    int idx = blockIdx.x * blockDim.x + threadIdx.x;
    if (idx >= 64 * 64 * 64) return;
    int c = idx >> 12;          // channel
    int p = idx & 4095;         // y*64 + x
    float v = sp[(size_t)p * 64 + c];
    out[idx] = 1.0f / (1.0f + __expf(-v));
}

// ---------------------------------------------------------------------------
// Host orchestration
// ---------------------------------------------------------------------------
extern "C" void kernel_launch(void* const* d_in, const int* in_sizes, int n_in,
                              void* d_out, int out_size, void* d_ws, size_t ws_size,
                              hipStream_t stream)
{
    (void)in_sizes; (void)n_in; (void)out_size; (void)ws_size;
    const float* x = (const float*)d_in[0];
    auto F = [&](int i) { return (const float*)d_in[i]; };

    float* ws = (float*)d_ws;
    size_t off = 0;
    auto alloc = [&](size_t nelem) { float* p = ws + off; off += nelem; return p; };

    const int    Mi = 65536;
    const size_t M  = 65536;

    float* htok = alloc(M * 64);
    float* lnb  = alloc(M * 64);
    float* xzB  = alloc(M * 256);
    float* xcc  = alloc(M * 128);
    float* dbl  = alloc(M * 36);
    float* dtb  = alloc(M * 128);
    float* ybuf = alloc(M * 128);   // also reused as MLP hidden
    float* g1   = alloc(16 * 64 * 64);
    float* g2   = alloc(64 * 32 * 32);
    float* g3   = alloc(64 * 16 * 16);
    float* g4   = alloc(64 * 16 * 16);
    float* gam  = alloc(96 * 16 * 16);
    float* bet  = alloc(96 * 16 * 16);
    float* xlow = alloc(1024 * 64);
    float* pool = alloc(1024 * 64);
    float* lnp  = alloc(1024 * 64);
    float* feat = alloc(1024 * 96);
    float* fchw = alloc(96 * 32 * 32);
    float* spec = alloc(96 * 16 * 16);
    float* lat  = alloc(96 * 16 * 16);
    float* dec0 = alloc(96 * 16 * 16);
    float* t1o  = alloc(96 * 32 * 32);
    float* t2o  = alloc(48 * 64 * 64);
    float* zc3  = alloc(64 * 64 * 64);
    float* sp   = alloc(4096 * 64);

    auto conv = [&](const float* in, const float* w, const float* bias, float* out,
                    int Cin, int Hin, int Win, int Cout, int Hout, int Wout,
                    int Kk, int st, int pad, int act) {
        int total = Cout * Hout * Wout;
        conv2d_k<<<CEILDIV(total, 256), 256, 0, stream>>>(
            in, w, bias, out, Cin, Hin, Win, Cout, Hout, Wout, Kk, st, pad, act);
    };
    auto gemm = [&](const float* A, const float* W, const float* bias, float* C,
                    int Mm, int N, int K, int lda, int ldc, int act, int accum) {
        dim3 grid(Mm / 16, CEILDIV(N, 64));
        dim3 blk(32, 4);
        if (K == 64)
            gemm_wmma_k<64><<<grid, blk, 0, stream>>>(A, W, bias, C, N, lda, ldc, act, accum);
        else
            gemm_wmma_k<128><<<grid, blk, 0, stream>>>(A, W, bias, C, N, lda, ldc, act, accum);
    };

    // --- spatial conditioning branch -> gamma, beta ---
    conv(x,  F(1), nullptr, g1, 64, 64, 64, 16, 64, 64, 1, 1, 0, ACT_GELU);
    conv(g1, F(2), nullptr, g2, 16, 64, 64, 64, 32, 32, 3, 2, 1, ACT_GELU);
    conv(g2, F(3), nullptr, g3, 64, 32, 32, 64, 16, 16, 3, 2, 1, ACT_GELU);
    conv(g3, F(4), nullptr, g4, 64, 16, 16, 64, 16, 16, 3, 1, 1, ACT_GELU);
    conv(g4, F(5), F(6), gam, 64, 16, 16, 96, 16, 16, 1, 1, 0, ACT_NONE);
    conv(g4, F(7), F(8), bet, 64, 16, 16, 96, 16, 16, 1, 1, 0, ACT_NONE);

    // --- tokens ---
    avgpool_k<<<CEILDIV(1024 * 64, 256), 256, 0, stream>>>(x, xlow);
    embed_k<<<CEILDIV((int)(M * 64), 256), 256, 0, stream>>>(xlow, F(9), F(10), F(11), htok);

    // --- 3x (bi-mamba + MLP) ---
    for (int b = 0; b < 3; ++b) {
        int B0 = 12 + 26 * b;
        ln_k<<<CEILDIV(Mi, 256), 256, 0, stream>>>(htok, F(B0 + 0), F(B0 + 1), lnb, Mi);
        for (int dir = 0; dir < 2; ++dir) {
            int P = B0 + 2 + 9 * dir;  // fwd / bwd param base
            gemm(lnb, F(P + 0), nullptr, xzB, Mi, 256, 64, 64, 256, ACT_NONE, 0);   // in_proj
            dwconv_silu_k<<<CEILDIV((int)(M * 128), 256), 256, 0, stream>>>(
                xzB, F(P + 1), F(P + 2), xcc, dir);
            gemm(xcc, F(P + 3), nullptr, dbl, Mi, 36, 128, 128, 36, ACT_NONE, 0);   // x_proj
            dtproj_k<<<CEILDIV((int)(M * 128), 256), 256, 0, stream>>>(
                dbl, F(P + 4), F(P + 5), dtb);
            scan_k<<<CEILDIV(1024 * 128, 256), 256, 0, stream>>>(
                dbl, dtb, xcc, xzB, F(P + 6), F(P + 7), ybuf, dir);
            gemm(ybuf, F(P + 8), nullptr, htok, Mi, 64, 128, 128, 64, ACT_NONE, 1); // out_proj (+residual)
        }
        // MLP
        ln_k<<<CEILDIV(Mi, 256), 256, 0, stream>>>(htok, F(B0 + 20), F(B0 + 21), lnb, Mi);
        gemm(lnb,  F(B0 + 22), F(B0 + 23), ybuf, Mi, 128, 64, 64, 128, ACT_GELU, 0);
        gemm(ybuf, F(B0 + 24), F(B0 + 25), htok, Mi, 64, 128, 128, 64, ACT_NONE, 1);
    }

    // --- final norm, attention pool, projection ---
    ln_k<<<CEILDIV(Mi, 256), 256, 0, stream>>>(htok, F(90), F(91), lnb, Mi);
    softmax_pool_k<<<1024, 64, 0, stream>>>(lnb, F(92), pool);
    ln_k<<<CEILDIV(1024, 256), 256, 0, stream>>>(pool, F(93), F(94), lnp, 1024);
    gemm(lnp, F(95), F(96), feat, 1024, 96, 64, 64, 96, ACT_GELU, 0);
    feat2chw_k<<<CEILDIV(96 * 1024, 256), 256, 0, stream>>>(feat, fchw);

    // --- spectral head + FiLM ---
    conv(fchw, F(97), F(98), spec, 96, 32, 32, 96, 16, 16, 2, 2, 0, ACT_NONE);
    film_k<<<CEILDIV(96 * 256, 256), 256, 0, stream>>>(spec, gam, bet);
    conv(spec, F(99), F(100), lat, 96, 16, 16, 96, 16, 16, 1, 1, 0, ACT_NONE);

    // --- decoder ---
    conv(lat, F(101), F(102), dec0, 96, 16, 16, 96, 16, 16, 1, 1, 0, ACT_GELU);
    convt_k<<<CEILDIV(96 * 32 * 32, 256), 256, 0, stream>>>(
        dec0, F(103), F(104), t1o, 96, 16, 16, 96, ACT_GELU);
    convt_k<<<CEILDIV(48 * 64 * 64, 256), 256, 0, stream>>>(
        t1o, F(105), F(106), t2o, 96, 32, 32, 48, ACT_GELU);
    conv(t2o, F(107), F(108), zc3, 48, 64, 64, 64, 64, 64, 3, 1, 1, ACT_NONE);

    // --- refinement + output ---
    chw2sp_k<<<CEILDIV(64 * 64 * 64, 256), 256, 0, stream>>>(zc3, sp);
    for (int r = 0; r < 3; ++r) {
        int R = 109 + 6 * r;
        ref_k<<<4096, 64, 0, stream>>>(sp, F(R), F(R + 1), F(R + 2), F(R + 3), F(R + 4), F(R + 5));
    }
    final_k<<<CEILDIV(64 * 64 * 64, 256), 256, 0, stream>>>(sp, (float*)d_out);
}
